// neural_network_2284922601909
// MI455X (gfx1250) — matrix operand order repaired and checked by reading, not compiled, not run on hardware
//
#include <hip/hip_runtime.h>
#include <hip/hip_bf16.h>

typedef __bf16 v16bf __attribute__((ext_vector_type(16)));
typedef float  v8f   __attribute__((ext_vector_type(8)));
typedef unsigned short u16;
typedef unsigned int   u32;

// float -> bf16 round-to-nearest-even
__device__ __forceinline__ u16 f2bf(float f) {
    u32 u = __float_as_uint(f);
    u32 r = (u + 0x7FFFu + ((u >> 16) & 1u)) >> 16;
    return (u16)r;
}

// Build a 16-element bf16 fragment from two 16-byte chunks.
__device__ __forceinline__ v16bf ld_frag(const u16* p0, const u16* p1) {
    union { v16bf v; uint4 q[2]; } u;
    u.q[0] = *(const uint4*)p0;
    u.q[1] = *(const uint4*)p1;
    return u.v;
}

// ---------------------------------------------------------------------------
// Kernel 0: weight prep.
//   Wt[t][oc][c]  = bf16( w2[oc][c][ky][kx] ),  t = ky*3+kx   (9*64*32)
//   Wf[n][k]      = bf16( fc1_w[n][k] )                        (128*9216)
// ---------------------------------------------------------------------------
__global__ void prep_weights(const float* __restrict__ w2,
                             const float* __restrict__ fc1_w,
                             u16* __restrict__ Wt, u16* __restrict__ Wf) {
    int i = blockIdx.x * blockDim.x + threadIdx.x;
    int stride = gridDim.x * blockDim.x;
    for (int j = i; j < 9 * 64 * 32; j += stride) {
        int t = j >> 11;            // /(64*32)
        int rem = j & 2047;
        int oc = rem >> 5;
        int c  = rem & 31;
        Wt[j] = f2bf(w2[oc * 288 + c * 9 + t]);
    }
    for (int j = i; j < 128 * 9216; j += stride)
        Wf[j] = f2bf(fc1_w[j]);
}

// ---------------------------------------------------------------------------
// Kernel 1: fused conv1+ReLU -> conv2(WMMA)+ReLU+2x2maxpool -> P (bf16).
// One workgroup per image. 256 threads = 8 wave32.
// LDS layout (dynamic, 84800 B):
//   xs[784] f32 | w1s[288] f32 | b1s[32] f32 | b2s[64] f32 |
//   h1[26*26*32] bf16 (y,x,c) | wt[9*64*32] bf16
// conv2 M-dim remap: m = q*4 + d, q = pooled index (oy*12+ox), d = 2x2 pos.
// ---------------------------------------------------------------------------
__global__ void conv_fused(const float* __restrict__ x,
                           const float* __restrict__ w1,
                           const float* __restrict__ b1,
                           const float* __restrict__ b2,
                           const u16* __restrict__ Wt,
                           u16* __restrict__ P) {
    extern __shared__ char smem[];
    float* xs  = (float*)smem;          // 784
    float* w1s = xs + 784;              // 288
    float* b1s = w1s + 288;             // 32
    float* b2s = b1s + 32;              // 64
    u16*   h1  = (u16*)(b2s + 64);      // 21632 bf16  (byte off 4672, 16B aligned)
    u16*   wt  = h1 + 26 * 26 * 32;     // 18432 bf16  (byte off 47936, 16B aligned)

    const int tid = threadIdx.x;
    const int img = blockIdx.x;

    // --- stage inputs into LDS ---
    const float* xp = x + (size_t)img * 784;
    for (int i = tid; i < 784; i += 256) xs[i] = xp[i];
    for (int i = tid; i < 288; i += 256) w1s[i] = w1[i];
    if (tid < 32) b1s[tid] = b1[tid];
    if (tid < 64) b2s[tid] = b2[tid];
    {
        const u32* s = (const u32*)Wt;
        u32* d = (u32*)wt;
        for (int i = tid; i < 9216; i += 256) d[i] = s[i];
    }
    __syncthreads();

    // --- conv1 + ReLU (VALU; 1.6% of FLOPs) -> h1[y][x][c] bf16 ---
    for (int p = tid; p < 676; p += 256) {
        int y = p / 26, xc = p % 26;
        float patch[9];
#pragma unroll
        for (int ky = 0; ky < 3; ++ky)
#pragma unroll
            for (int kx = 0; kx < 3; ++kx)
                patch[ky * 3 + kx] = xs[(y + ky) * 28 + (xc + kx)];
        u16* dst = h1 + p * 32;
        for (int c = 0; c < 32; ++c) {
            float acc = b1s[c];
            const float* wr = w1s + c * 9;
#pragma unroll
            for (int j = 0; j < 9; ++j) acc = fmaf(wr[j], patch[j], acc);
            dst[c] = f2bf(fmaxf(acc, 0.0f));
        }
    }
    __syncthreads();

    // --- conv2 as implicit GEMM: M=576 (remapped), N=64, K=288 (9 taps x 32ch)
    const int wv = tid >> 5;
    const int l  = tid & 31;
    const int lm = l & 15;
    const int aSel = (l & 16) ? 8 : 0;    // A: K half select {0..7,16..23}/{8..15,24..31}
    const int bSel = (l & 16) ? 8 : 0;    // B: K half select {0..7,16..23}/{8..15,24..31}

    for (int tile = wv; tile < 144; tile += 8) {   // 36 M-tiles x 4 N-tiles
        const int tm = tile >> 2, tn = tile & 3;
        // lane geometry: row m -> (pooled q, window pos d) -> conv2 out (yy,xx)
        const int m  = tm * 16 + lm;
        const int q  = m >> 2, d = m & 3;
        const int oy = q / 12, ox = q % 12;
        const int yy = 2 * oy + (d >> 1);
        const int xx = 2 * ox + (d & 1);
        const u16* abase = h1 + (yy * 26 + xx) * 32 + aSel;
        const int  n     = tn * 16 + lm;
        const u16* bbase = wt + n * 32 + bSel;

        const float bias = b2s[n];
        v8f acc = {bias, bias, bias, bias, bias, bias, bias, bias};
#pragma unroll
        for (int t = 0; t < 9; ++t) {
            const int ky = t / 3, kx = t % 3;
            const u16* ap = abase + (ky * 26 + kx) * 32;
            const u16* bp = bbase + t * 64 * 32;
            v16bf av = ld_frag(ap, ap + 16);
            v16bf bv = ld_frag(bp, bp + 16);
            acc = __builtin_amdgcn_wmma_f32_16x16x32_bf16(
                false, av, false, bv, (short)0, acc, false, false);
        }
        // ReLU + 2x2 maxpool entirely in the accumulator:
        // lanes 0-15: v0..3 -> q=tm*4+0, v4..7 -> q=tm*4+1
        // lanes16-31: v0..3 -> q=tm*4+2, v4..7 -> q=tm*4+3
        float p0 = fmaxf(fmaxf(acc[0], acc[1]), fmaxf(acc[2], acc[3]));
        float p1 = fmaxf(fmaxf(acc[4], acc[5]), fmaxf(acc[6], acc[7]));
        p0 = fmaxf(p0, 0.0f);
        p1 = fmaxf(p1, 0.0f);
        const int qa = tm * 4 + ((l & 16) ? 2 : 0);
        u16* prow = P + (size_t)img * 9216 + (size_t)n * 144;  // NCHW flatten: c*144+q
        prow[qa]     = f2bf(p0);
        prow[qa + 1] = f2bf(p1);
    }
}

// ---------------------------------------------------------------------------
// Kernel 2: fc1 GEMM  H[4096,128] = relu(P[4096,9216] @ Wf^T + b)  via WMMA.
// 2048 tiles of 16x16; one tile per wave; K loop = 288 steps of 32.
// ---------------------------------------------------------------------------
__global__ void fc1_wmma(const u16* __restrict__ P,
                         const u16* __restrict__ Wf,
                         const float* __restrict__ fc1_b,
                         float* __restrict__ H) {
    const int tid = threadIdx.x;
    const int wv = tid >> 5;
    const int l  = tid & 31;
    const int lm = l & 15;
    const int tile = blockIdx.x * 8 + wv;   // 0..2047
    const int tm = tile >> 3;               // 0..255  (M tile, 16 images)
    const int tn = tile & 7;                // 0..7    (N tile, 16 outputs)
    const int aSel = (l & 16) ? 8 : 0;
    const int bSel = (l & 16) ? 8 : 0;

    const u16* arow = P  + (size_t)(tm * 16 + lm) * 9216 + aSel;
    const int  n    = tn * 16 + lm;
    const u16* brow = Wf + (size_t)n * 9216 + bSel;

    const float bias = fc1_b[n];
    v8f acc = {bias, bias, bias, bias, bias, bias, bias, bias};
#pragma unroll 4
    for (int k0 = 0; k0 < 9216; k0 += 32) {
        v16bf av = ld_frag(arow + k0, arow + k0 + 16);
        v16bf bv = ld_frag(brow + k0, brow + k0 + 16);
        acc = __builtin_amdgcn_wmma_f32_16x16x32_bf16(
            false, av, false, bv, (short)0, acc, false, false);
    }
    const int mb = tm * 16 + ((l & 16) ? 8 : 0);
#pragma unroll
    for (int r = 0; r < 8; ++r)
        H[(size_t)(mb + r) * 128 + n] = fmaxf(acc[r], 0.0f);
}

// ---------------------------------------------------------------------------
// Kernel 3: fc2  out[4096,10] = H @ fc2_w^T + b  (tiny; VALU)
// ---------------------------------------------------------------------------
__global__ void fc2_plain(const float* __restrict__ H,
                          const float* __restrict__ fc2_w,
                          const float* __restrict__ fc2_b,
                          float* __restrict__ out) {
    int idx = blockIdx.x * blockDim.x + threadIdx.x;
    if (idx >= 4096 * 10) return;
    int img = idx / 10, cls = idx % 10;
    const float* h = H + (size_t)img * 128;
    const float* w = fc2_w + (size_t)cls * 128;
    float acc = fc2_b[cls];
#pragma unroll 8
    for (int k = 0; k < 128; ++k) acc = fmaf(h[k], w[k], acc);
    out[idx] = acc;
}

// ---------------------------------------------------------------------------
extern "C" void kernel_launch(void* const* d_in, const int* in_sizes, int n_in,
                              void* d_out, int out_size, void* d_ws, size_t ws_size,
                              hipStream_t stream) {
    const float* x     = (const float*)d_in[0];
    const float* w1    = (const float*)d_in[1];
    const float* b1    = (const float*)d_in[2];
    const float* w2    = (const float*)d_in[3];
    const float* b2    = (const float*)d_in[4];
    const float* fc1_w = (const float*)d_in[5];
    const float* fc1_b = (const float*)d_in[6];
    const float* fc2_w = (const float*)d_in[7];
    const float* fc2_b = (const float*)d_in[8];
    float* out = (float*)d_out;

    // workspace layout (all 256B aligned)
    char* ws = (char*)d_ws;
    u16*   P  = (u16*)ws;                                   // 4096*9216 bf16 = 75,497,472 B
    u16*   Wt = (u16*)(ws + 75497472);                      // 9*64*32 bf16   =     36,864 B
    u16*   Wf = (u16*)(ws + 75497472 + 36864);              // 128*9216 bf16  =  2,359,296 B
    float* H  = (float*)(ws + 75497472 + 36864 + 2359296);  // 4096*128 f32   =  2,097,152 B

    prep_weights<<<1024, 256, 0, stream>>>(w2, fc1_w, Wt, Wf);
    conv_fused<<<4096, 256, 84800, stream>>>(x, w1, b1, b2, Wt, P);
    fc1_wmma<<<256, 256, 0, stream>>>(P, Wf, fc1_b, H);
    fc2_plain<<<160, 256, 0, stream>>>(H, fc2_w, fc2_b, out);
}
